// CircularConvolutionalAttention_22840636080441
// MI455X (gfx1250) — matrix operand order repaired and checked by reading, not compiled, not run on hardware
//
#include <hip/hip_runtime.h>

// ---------------------------------------------------------------------------
// CircularConvolutionalAttention on MI455X (gfx1250, wave32, WMMA)
//
//   z      = x @ W_A                 (B,h,N)   small, VALU
//   w      = softmax_N(z) / N        (B,h,N)   stored bf16
//   V      = x @ W_V                 (B,N,768) bf16 WMMA GEMM
//   mid    = circulant(w) @ V        (B,N,768) bf16 WMMA, A = LDS sliding window
//   out    = mid @ W_proj + b        (B,N,768) f32  WMMA GEMM
//
// All matrix math uses v_wmma_f32_16x16x32_bf16 (bf16 in, f32 accumulate).
// Fragment construction matches the ISA 7.12.2 VGPR mappings with wide LDS
// loads (2x ds_load_b128 per fragment; 8x ds_load_b32 for the unaligned
// sliding-window A of the circulant). GEMM dims are compile-time so all
// address math is 32-bit constant-stride.
// ---------------------------------------------------------------------------

typedef __attribute__((ext_vector_type(16))) __bf16 v16bf;
typedef __attribute__((ext_vector_type(8)))  float  v8f;

#define B_  4
#define N_  2048
#define C_  768
#define H_  12
#define HD_ 64
#define GM_ (B_ * N_)   // 8192 rows in both big GEMMs

union FragQ { uint4 q[2]; v16bf v; };        // 2x 16B chunks -> one fragment
union FragU { uint32_t u[8]; v16bf v; };     // 8x dwords     -> one fragment

// ---------------------------------------------------------------- cast f32->bf16
__global__ void cast_f32_bf16_kernel(const float* __restrict__ src,
                                     __bf16* __restrict__ dst, int n) {
  int i = (blockIdx.x * blockDim.x + threadIdx.x) * 4;
  if (i < n) {
    float4 v = *(const float4*)(src + i);
    dst[i + 0] = (__bf16)v.x;
    dst[i + 1] = (__bf16)v.y;
    dst[i + 2] = (__bf16)v.z;
    dst[i + 3] = (__bf16)v.w;
  }
}

// ---------------------------------------------------------------- router logits
__global__ __launch_bounds__(256)
void router_kernel(const float* __restrict__ x, const float* __restrict__ WA,
                   float* __restrict__ z) {
  __shared__ float WAs[C_ * H_];
  for (int t = threadIdx.x; t < C_ * H_; t += blockDim.x) WAs[t] = WA[t];
  __syncthreads();

  int row = blockIdx.x * blockDim.x + threadIdx.x;   // exact grid: B*N threads
  const float* xr = x + row * C_;
  float acc[H_];
#pragma unroll
  for (int h = 0; h < H_; ++h) acc[h] = 0.f;
  for (int c = 0; c < C_; ++c) {
    float xv = xr[c];
#pragma unroll
    for (int h = 0; h < H_; ++h) acc[h] += xv * WAs[c * H_ + h];
  }
  int b = row / N_, n = row % N_;
#pragma unroll
  for (int h = 0; h < H_; ++h) z[(b * H_ + h) * N_ + n] = acc[h];
}

// ---------------------------------------------------------------- softmax / N
__global__ __launch_bounds__(256)
void softmax_kernel(const float* __restrict__ z, __bf16* __restrict__ wbf) {
  int bh = blockIdx.x;
  const float* zr = z + bh * N_;
  __shared__ float red[256];
  int tid = threadIdx.x;

  float m = -1e30f;
  for (int i = tid; i < N_; i += 256) m = fmaxf(m, zr[i]);
  red[tid] = m; __syncthreads();
  for (int s = 128; s > 0; s >>= 1) {
    if (tid < s) red[tid] = fmaxf(red[tid], red[tid + s]);
    __syncthreads();
  }
  m = red[0]; __syncthreads();

  float sum = 0.f;
  for (int i = tid; i < N_; i += 256) sum += __expf(zr[i] - m);
  red[tid] = sum; __syncthreads();
  for (int s = 128; s > 0; s >>= 1) {
    if (tid < s) red[tid] += red[tid + s];
    __syncthreads();
  }
  float inv = 1.0f / (red[0] * (float)N_);    // fold 1/N into w
  for (int i = tid; i < N_; i += 256)
    wbf[bh * N_ + i] = (__bf16)(__expf(zr[i] - m) * inv);
}

// ---------------------------------------------------------------- bf16 WMMA GEMM
// C[GM_,768] = A[GM_,768] * B[768,768]; 64x64 tile, K-step 32, 8 waves.
// Compile-time dims -> 32-bit constant-stride addressing throughout.
__global__ __launch_bounds__(256)
void gemm_bf16_wmma(const __bf16* __restrict__ A, const __bf16* __restrict__ Bm,
                    __bf16* __restrict__ Cbf, float* __restrict__ Cf,
                    const float* __restrict__ bias) {
  __shared__ uint4 As4[64][5];     // row r: k-chunk q -> k = 8q..8q+7 (pad=1)
  __shared__ uint4 BsT4[64][5];    // col n: k-chunk q -> k = 8q..8q+7 (pad=1)

  int tid  = threadIdx.x;
  int wave = tid >> 5, lane = tid & 31;
  int Mloc = lane & 15, g = lane >> 4;
  int i0 = blockIdx.y * 64;
  int n0 = blockIdx.x * 64;
  int it  = wave & 3;                       // i-tile 0..3
  int ntb = (wave >> 2) * 2;                // n-tile base: 0 or 2

  // staging coordinates (constant per thread)
  int ar = tid >> 2, aq = tid & 3;          // A: one uint4 per thread
  int br = tid >> 5, bc2 = (tid & 31) * 2;  // B: 4 uints per thread (r += 8)

  v8f acc0 = {}; v8f acc1 = {};

  for (int k0 = 0; k0 < C_; k0 += 32) {
    // stage A: 64 rows x 4 uint4 chunks, one global_load_b128 per thread
    As4[ar][aq] = *(const uint4*)(A + (i0 + ar) * C_ + k0 + 8 * aq);
    // stage B transposed: rows k0+br+8p, scatter u16 pairs to BsT
#pragma unroll
    for (int p = 0; p < 4; ++p) {
      int r = br + 8 * p;
      uint32_t v = *(const uint32_t*)(Bm + (k0 + r) * C_ + n0 + bc2);
      union { uint32_t w; __bf16 h[2]; } cv; cv.w = v;
      ((__bf16*)BsT4[bc2])[r]     = cv.h[0];
      ((__bf16*)BsT4[bc2 + 1])[r] = cv.h[1];
    }
    __syncthreads();

    FragQ fa;                                // A: 2x ds_load_b128
    fa.q[0] = As4[it * 16 + Mloc][g];
    fa.q[1] = As4[it * 16 + Mloc][2 + g];
    FragQ fb0, fb1;                          // B: 2x ds_load_b128 each
    fb0.q[0] = BsT4[ntb * 16 + Mloc][g];
    fb0.q[1] = BsT4[ntb * 16 + Mloc][2 + g];
    fb1.q[0] = BsT4[(ntb + 1) * 16 + Mloc][g];
    fb1.q[1] = BsT4[(ntb + 1) * 16 + Mloc][2 + g];

    acc0 = __builtin_amdgcn_wmma_f32_16x16x32_bf16(false, fa.v, false, fb0.v,
                                                   (short)0, acc0, false, false);
    acc1 = __builtin_amdgcn_wmma_f32_16x16x32_bf16(false, fa.v, false, fb1.v,
                                                   (short)0, acc1, false, false);
    __syncthreads();
  }

  int col0 = n0 + ntb * 16 + Mloc;
  int col1 = col0 + 16;
  int rbase = (i0 + it * 16 + 8 * g) * C_;
  if (Cf) {
    float b0 = bias ? bias[col0] : 0.f;
    float b1 = bias ? bias[col1] : 0.f;
#pragma unroll
    for (int r = 0; r < 8; ++r) {
      Cf[rbase + r * C_ + col0] = acc0[r] + b0;
      Cf[rbase + r * C_ + col1] = acc1[r] + b1;
    }
  } else {
#pragma unroll
    for (int r = 0; r < 8; ++r) {
      Cbf[rbase + r * C_ + col0] = (__bf16)acc0[r];
      Cbf[rbase + r * C_ + col1] = (__bf16)acc1[r];
    }
  }
}

// ---------------------------------------------------------------- circulant WMMA
// mid[b, i, h*64+d] = sum_j w[b,h,(j-i) mod N] * V[b, j, h*64+d]
// A-tile elements come from a sliding window over w held in LDS. The window
// base has per-lane parity, so we keep two dword-packed copies:
//   WlA[t] = {w[2t], w[2t+1]},  WlB[t] = {w[2t+1], w[2t+2]}
// and each lane reads 8 aligned dwords (parity-matched) per fragment.
__global__ __launch_bounds__(128)
void circconv_wmma(const __bf16* __restrict__ wbf, const __bf16* __restrict__ V,
                   __bf16* __restrict__ mid) {
  __shared__ uint32_t WlA[1056];   // 2112 bf16 window: N + 64 wrap margin
  __shared__ uint32_t WlB[1056];   // same window shifted by one element
  __shared__ uint4 VsT4[64][5];    // V tile transposed: col d, k-chunks

  int bh = blockIdx.x;             // 0..47
  int b = bh / H_, h = bh % H_;
  int i0 = blockIdx.y * 64;
  int tid  = threadIdx.x;
  int wave = tid >> 5, lane = tid & 31;
  int Mloc = lane & 15, g = lane >> 4;
  int iloc = wave * 16 + Mloc;     // local output row within the 64-row block

  // stage w window: Wl[s] = w[(s - i0 - 63) mod N]
  int shift = (i0 + 63) & (N_ - 1);
  const __bf16* wr = wbf + bh * N_;
  __bf16* Wl = (__bf16*)WlA;
  for (int s = tid; s < 2112; s += 128)
    Wl[s] = wr[(s + N_ - shift) & (N_ - 1)];
  __syncthreads();
  for (int t = tid; t < 1056; t += 128) {
    union { uint32_t w; __bf16 h[2]; } cv;
    cv.h[0] = Wl[2 * t + 1];
    cv.h[1] = (2 * t + 2 < 2112) ? Wl[2 * t + 2] : (__bf16)0.f;
    WlB[t] = cv.w;
  }

  // per-lane window pointer (parity fixed across the whole K loop)
  int base0 = 63 - iloc + 8 * g;           // >= 0
  const uint32_t* wp = (base0 & 1) ? (WlB + ((base0 - 1) >> 1))
                                   : (WlA + (base0 >> 1));

  // staging coordinates for the V tile (constant per thread)
  int vr = tid >> 5, vc2 = (tid & 31) * 2;  // 8 rows per thread (r += 4)
  int vbase = (b * N_ + vr) * C_ + h * HD_ + vc2;   // + j0*C_ per step

  v8f acc[4];
#pragma unroll
  for (int nt = 0; nt < 4; ++nt) acc[nt] = (v8f){};

  for (int j0 = 0; j0 < N_; j0 += 32) {
    __syncthreads();
    // stage V[j0..j0+31, h*64..h*64+63] transposed -> VsT
#pragma unroll
    for (int p = 0; p < 8; ++p) {
      int r = vr + 4 * p;
      uint32_t v = *(const uint32_t*)(V + vbase + (j0 + 4 * p) * C_);
      union { uint32_t w; __bf16 hh[2]; } cv; cv.w = v;
      ((__bf16*)VsT4[vc2])[r]     = cv.hh[0];
      ((__bf16*)VsT4[vc2 + 1])[r] = cv.hh[1];
    }
    // prefetch next K-step tile (global_prefetch_b8)
    if (j0 + 32 < N_) {
      const char* np =
          (const char*)(V + ((b * N_ + j0 + 32 + (tid >> 2)) * C_ + h * HD_));
      __builtin_prefetch(np, 0, 1);
    }
    __syncthreads();

    // A fragment: 8 aligned dword loads from the parity-matched window copy
    FragU fa;
    int idx = j0 >> 1;
#pragma unroll
    for (int m2 = 0; m2 < 4; ++m2) {
      fa.u[m2]     = wp[idx + m2];         // elems e=2m2, 2m2+1   (k run 1)
      fa.u[4 + m2] = wp[idx + 8 + m2];     // elems e=8+2m2, ...   (k run 2)
    }
#pragma unroll
    for (int nt = 0; nt < 4; ++nt) {
      FragQ fb;                             // 2x ds_load_b128
      fb.q[0] = VsT4[nt * 16 + Mloc][g];
      fb.q[1] = VsT4[nt * 16 + Mloc][2 + g];
      acc[nt] = __builtin_amdgcn_wmma_f32_16x16x32_bf16(false, fa.v, false, fb.v,
                                                        (short)0, acc[nt],
                                                        false, false);
    }
  }

  int obase = (b * N_ + i0 + wave * 16 + 8 * g) * C_ + h * HD_ + Mloc;
#pragma unroll
  for (int nt = 0; nt < 4; ++nt) {
#pragma unroll
    for (int r = 0; r < 8; ++r) {
      mid[obase + r * C_ + nt * 16] = (__bf16)acc[nt][r];
    }
  }
}

// ---------------------------------------------------------------- launch
extern "C" void kernel_launch(void* const* d_in, const int* in_sizes, int n_in,
                              void* d_out, int out_size, void* d_ws, size_t ws_size,
                              hipStream_t stream) {
  const float* x  = (const float*)d_in[0];
  const float* WA = (const float*)d_in[1];
  const float* WV = (const float*)d_in[2];
  const float* WP = (const float*)d_in[3];
  const float* bp = (const float*)d_in[4];
  float* out = (float*)d_out;

  char* ws = (char*)d_ws;
  size_t off = 0;
  auto alloc = [&](size_t bytes) -> void* {
    void* p = ws + off;
    off = (off + bytes + 255) & ~(size_t)255;
    return p;
  };
  __bf16* xbf   = (__bf16*)alloc((size_t)GM_ * C_ * 2);
  __bf16* wvbf  = (__bf16*)alloc((size_t)C_ * C_ * 2);
  __bf16* wpbf  = (__bf16*)alloc((size_t)C_ * C_ * 2);
  __bf16* vbf   = (__bf16*)alloc((size_t)GM_ * C_ * 2);
  __bf16* midbf = (__bf16*)alloc((size_t)GM_ * C_ * 2);
  float*  z     = (float*) alloc((size_t)B_ * H_ * N_ * 4);
  __bf16* wbf   = (__bf16*)alloc((size_t)B_ * H_ * N_ * 2);

  int nx = GM_ * C_;
  int nw = C_ * C_;
  cast_f32_bf16_kernel<<<(nx / 4 + 255) / 256, 256, 0, stream>>>(x,  xbf,  nx);
  cast_f32_bf16_kernel<<<(nw / 4 + 255) / 256, 256, 0, stream>>>(WV, wvbf, nw);
  cast_f32_bf16_kernel<<<(nw / 4 + 255) / 256, 256, 0, stream>>>(WP, wpbf, nw);

  router_kernel<<<GM_ / 256, 256, 0, stream>>>(x, WA, z);
  softmax_kernel<<<B_ * H_, 256, 0, stream>>>(z, wbf);

  // V = x @ W_V  (bf16 out)
  gemm_bf16_wmma<<<dim3(C_ / 64, GM_ / 64), 256, 0, stream>>>(
      xbf, wvbf, vbf, nullptr, nullptr);

  // mid = circulant(w) @ V
  circconv_wmma<<<dim3(B_ * H_, N_ / 64), 128, 0, stream>>>(wbf, vbf, midbf);

  // out = mid @ W_proj + b  (f32 out)
  gemm_bf16_wmma<<<dim3(C_ / 64, GM_ / 64), 256, 0, stream>>>(
      midbf, wpbf, nullptr, out, bp);
}
